// NodeUpdateLayer_75831942578377
// MI455X (gfx1250) — matrix operand order repaired and checked by reading, not compiled, not run on hardware
//
#include <hip/hip_runtime.h>
#include <hip/hip_bf16.h>

#define NN 10000
#define EE 640000

typedef __attribute__((ext_vector_type(16))) _Float16 v16h;
typedef __attribute__((ext_vector_type(8)))  float    v8f;

__device__ __forceinline__ v8f wmma_f16(v16h a, v16h b, v8f c) {
  return __builtin_amdgcn_wmma_f32_16x16x32_f16(false, a, false, b, (short)0, c,
                                                false, false);
}

// hardware tanh (V_TANH_F32, new on gfx1250) with portable fallback
__device__ __forceinline__ float htanh(float x) {
#if __has_builtin(__builtin_amdgcn_tanhf)
  return __builtin_amdgcn_tanhf(x);
#else
  return tanhf(x);
#endif
}

// sum across each group of 8 lanes via ds_swizzle_b32 xor patterns
// (group-of-32 mode: offset = xor_mask<<10 | and_mask(0x1f))
__device__ __forceinline__ float headsum8(float x) {
  x += __uint_as_float(
      (unsigned)__builtin_amdgcn_ds_swizzle((int)__float_as_uint(x), 0x041F));
  x += __uint_as_float(
      (unsigned)__builtin_amdgcn_ds_swizzle((int)__float_as_uint(x), 0x081F));
  x += __uint_as_float(
      (unsigned)__builtin_amdgcn_ds_swizzle((int)__float_as_uint(x), 0x101F));
  return x;
}

// ---------------------------------------------------------------------------
// Fragment-major ("swizzled") LDS layouts: each lane's 16-half WMMA fragment
// is contiguous (32 bytes) so it loads as ds_load_b128 x2, no packing movs.
//
// B (32x16 tile, same K order as A: lanes 0-15: K={0..7,16..23}; lanes 16-31: K={8..15,24..31}):
//   dst half-index = ((ktile*8 + n/16)*32 + ((n&15)|((k<<1)&16)))*16 + ((k&7)|((k>>1)&8))
// A (16x32 tile, ISA: lanes 0-15: K={0..7,16..23}; lanes 16-31: K={8..15,24..31}):
//   dst half-index = ((ktile*32 + (m | ((k<<1)&16)))*16) + ((k&7)|((k>>1)&8))
// ---------------------------------------------------------------------------
__device__ __forceinline__ void stage_b_sw(_Float16* Bs, const float* __restrict__ B,
                                           int K) {
  for (int i = threadIdx.x; i < K * 128; i += 256) {
    const int k = i >> 7, n = i & 127;
    const int idx = ((((k >> 5) << 3) | (n >> 4)) * 32 + ((n & 15) | ((k << 1) & 16))) * 16 +
                    ((k & 7) | ((k >> 1) & 8));
    Bs[idx] = (_Float16)B[i];
  }
}

__device__ __forceinline__ v16h b_frag_sw(const _Float16* Bs, int kt, int coltile,
                                          int lane) {
  return *(const v16h*)(Bs + (((kt << 3) | coltile) * 32 + lane) * 16);
}

__device__ __forceinline__ v16h a_frag_sw(const _Float16* As, int kt, int lane) {
  return *(const v16h*)(As + (kt * 32 + lane) * 16);
}

// A fragment built directly from row-major fp32 global memory (f32 -> f16 cvt)
__device__ __forceinline__ v16h a_frag_g(const float* __restrict__ A, int lda,
                                         int row0, int k0, int lane) {
  const int m = lane & 15;
  const int koff = (lane < 16) ? 0 : 8;
  const float* p = A + (size_t)(row0 + m) * lda + k0 + koff;
  v16h a;
#pragma unroll
  for (int j = 0; j < 8; ++j) a[j] = (_Float16)p[j];
#pragma unroll
  for (int j = 0; j < 8; ++j) a[j + 8] = (_Float16)p[j + 16];
  return a;
}

// order-preserving float -> uint key for atomicMax-based segment max
__device__ __forceinline__ unsigned fkey(float x) {
  unsigned u = __float_as_uint(x);
  return (u & 0x80000000u) ? ~u : (u | 0x80000000u);
}
__device__ __forceinline__ float funkey(unsigned k) {
  return __uint_as_float((k & 0x80000000u) ? (k ^ 0x80000000u) : ~k);
}

// ---------------- Generic GEMM: C[M,128] = A[M,128] @ B[128,128] (+bias) --------
template <bool BIAS>
__global__ void gemm_k128(const float* __restrict__ A, const float* __restrict__ B,
                          const float* __restrict__ bias, float* __restrict__ C) {
  __shared__ __align__(32) _Float16 Bs[128 * 128];
  stage_b_sw(Bs, B, 128);
  __syncthreads();
  const int wave = threadIdx.x >> 5, lane = threadIdx.x & 31;
  const int row0 = blockIdx.x * 16, col0 = wave * 16;
  v8f c = {};
#pragma unroll
  for (int kt = 0; kt < 4; ++kt)
    c = wmma_f16(a_frag_g(A, 128, row0, kt * 32, lane), b_frag_sw(Bs, kt, wave, lane), c);
  const int n = lane & 15, mb = (lane < 16) ? 0 : 8;
  const float bv = BIAS ? bias[col0 + n] : 0.0f;
#pragma unroll
  for (int r = 0; r < 8; ++r)
    C[(size_t)(row0 + mb + r) * 128 + col0 + n] = c[r] + bv;
}

// stage one 16x64 fp32 tile into LDS via the CDNA5 async-DMA path
// (exactly 256 lanes x 16B = 4KB in a single global_load_async_to_lds_b128).
// Generic pointers to __shared__ carry the wave-relative LDS offset in their
// low 32 bits (ISA 10.2 aperture mapping), which is what VDST expects.
__device__ __forceinline__ void async_stage_tile(float* lds, const float* __restrict__ g) {
  const unsigned ldsOff = (unsigned)(size_t)(lds + threadIdx.x * 4);
  const float* gp = g + threadIdx.x * 4;
  asm volatile("global_load_async_to_lds_b128 %0, %1, off"
               :: "v"(ldsOff), "v"(gp)
               : "memory");
  asm volatile("s_wait_asynccnt 0" ::: "memory");
}

// ---------- Pass A: e0 = tanh(ef@We0) fused with alpha + segment max ------------
__global__ void edge_alpha_kernel(const float* __restrict__ ef, const int* __restrict__ eidx,
                                  const float* __restrict__ We0, const float* __restrict__ q,
                                  const float* __restrict__ kmat, float* __restrict__ alpha,
                                  unsigned* __restrict__ amaxkey) {
  __shared__ __align__(32) _Float16 Bs[64 * 128];
  __shared__ __align__(32) _Float16 As[16 * 64];
  __shared__ __align__(16) float efStage[16 * 64];
  __shared__ int sSrc[16], sDst[16];
  const int e0b = blockIdx.x * 16;
  async_stage_tile(efStage, ef + (size_t)e0b * 64);  // DMA completes (s_wait_asynccnt 0) before weight staging
  stage_b_sw(Bs, We0, 64);
  if (threadIdx.x < 16) {
    sSrc[threadIdx.x] = eidx[e0b + threadIdx.x];
    sDst[threadIdx.x] = eidx[EE + e0b + threadIdx.x];
  }
  if (threadIdx.x == 0 && e0b + 16 < EE)
    __builtin_prefetch(ef + (size_t)(e0b + 16) * 64, 0, 1);
  __syncthreads();
  for (int i = threadIdx.x; i < 16 * 64; i += 256) {
    const int m = i >> 6, k = i & 63;
    As[(((k >> 5) * 32 + (m | ((k << 1) & 16))) * 16) + ((k & 7) | ((k >> 1) & 8))] =
        (_Float16)efStage[i];
  }
  __syncthreads();
  const int wave = threadIdx.x >> 5, lane = threadIdx.x & 31;
  v8f c = {};
#pragma unroll
  for (int kt = 0; kt < 2; ++kt)
    c = wmma_f16(a_frag_sw(As, kt, lane), b_frag_sw(Bs, kt, wave, lane), c);
  const int n = lane & 15, mb = (lane < 16) ? 0 : 8;
  const int colg = wave * 16 + n;
  const int h = colg >> 3;  // head index (C = 8)
#pragma unroll
  for (int r = 0; r < 8; ++r) {
    const int m = mb + r;
    const int s = sSrc[m], d = sDst[m];
    float val = headsum8(htanh(c[r]) * q[(size_t)d * 128 + colg] *
                         kmat[(size_t)s * 128 + colg]);
    if ((lane & 7) == 0) {
      const float a = val * 0.35355339059327373f;  // 1/sqrt(8)
      alpha[(size_t)(e0b + m) * 16 + h] = a;
      atomicMax(&amaxkey[(unsigned)d * 16 + h], fkey(a));
    }
  }
}

// ------------- softmax: expa in place + segment-sum denominator -----------------
__global__ void softmax_denom_kernel(const int* __restrict__ eidx,
                                     const unsigned* __restrict__ amaxkey,
                                     float* __restrict__ alpha, float* __restrict__ denom) {
  const int i = blockIdx.x * blockDim.x + threadIdx.x;  // over E*16 exactly
  const int e = i >> 4, h = i & 15;
  const int d = eidx[EE + e];
  const float mx = funkey(amaxkey[d * 16 + h]);
  const float ex = __expf(alpha[i] - mx);  // arg <= 0: v_exp_f32 path is ample
  alpha[i] = ex;
  atomicAdd(&denom[d * 16 + h], ex);
}

// ---- Pass B: e1 = tanh(ef@We1) fused with gated message scatter-add ------------
__global__ void edge_msg_kernel(const float* __restrict__ ef, const int* __restrict__ eidx,
                                const float* __restrict__ We1, const float* __restrict__ v,
                                const float* __restrict__ expa, const float* __restrict__ denom,
                                float* __restrict__ outb) {
  __shared__ __align__(32) _Float16 Bs[64 * 128];
  __shared__ __align__(32) _Float16 As[16 * 64];
  __shared__ __align__(16) float efStage[16 * 64];
  __shared__ int sSrc[16], sDst[16];
  const int e0b = blockIdx.x * 16;
  async_stage_tile(efStage, ef + (size_t)e0b * 64);
  stage_b_sw(Bs, We1, 64);
  if (threadIdx.x < 16) {
    sSrc[threadIdx.x] = eidx[e0b + threadIdx.x];
    sDst[threadIdx.x] = eidx[EE + e0b + threadIdx.x];
  }
  if (threadIdx.x == 0 && e0b + 16 < EE)
    __builtin_prefetch(ef + (size_t)(e0b + 16) * 64, 0, 1);
  __syncthreads();
  for (int i = threadIdx.x; i < 16 * 64; i += 256) {
    const int m = i >> 6, k = i & 63;
    As[(((k >> 5) * 32 + (m | ((k << 1) & 16))) * 16) + ((k & 7) | ((k >> 1) & 8))] =
        (_Float16)efStage[i];
  }
  __syncthreads();
  const int wave = threadIdx.x >> 5, lane = threadIdx.x & 31;
  v8f c = {};
#pragma unroll
  for (int kt = 0; kt < 2; ++kt)
    c = wmma_f16(a_frag_sw(As, kt, lane), b_frag_sw(Bs, kt, wave, lane), c);
  const int n = lane & 15, mb = (lane < 16) ? 0 : 8;
  const int colg = wave * 16 + n;
  const int h = colg >> 3;
#pragma unroll
  for (int r = 0; r < 8; ++r) {
    const int m = mb + r;
    const int s = sSrc[m], d = sDst[m];
    const int e = e0b + m;
    const float attn = expa[(size_t)e * 16 + h] / (denom[(size_t)d * 16 + h] + 1e-16f);
    const float msg = v[(size_t)s * 128 + colg] * htanh(c[r]) * attn;
    atomicAdd(&outb[(size_t)d * 128 + colg], msg);
  }
}

// ---- MLP1: h = relu(LN(cat(out,x)@W1 + b1)) — split-K WMMA + in-LDS LayerNorm --
__global__ void mlp1_kernel(const float* __restrict__ outb, const float* __restrict__ x,
                            const float* __restrict__ W1, const float* __restrict__ b1,
                            const float* __restrict__ ln_g, const float* __restrict__ ln_b,
                            float* __restrict__ hbuf) {
  __shared__ __align__(32) _Float16 Bs[128 * 128];
  __shared__ float hs[16][128];
  __shared__ float smu[16], srstd[16];
  const int wave = threadIdx.x >> 5, lane = threadIdx.x & 31;
  const int row0 = blockIdx.x * 16, col0 = wave * 16;
  v8f c = {};
  // K-half 1: A = outb, B = W1 rows [0,128)
  stage_b_sw(Bs, W1, 128);
  __syncthreads();
#pragma unroll
  for (int kt = 0; kt < 4; ++kt)
    c = wmma_f16(a_frag_g(outb, 128, row0, kt * 32, lane), b_frag_sw(Bs, kt, wave, lane), c);
  __syncthreads();
  // K-half 2: A = x, B = W1 rows [128,256)
  stage_b_sw(Bs, W1 + 128 * 128, 128);
  __syncthreads();
#pragma unroll
  for (int kt = 0; kt < 4; ++kt)
    c = wmma_f16(a_frag_g(x, 128, row0, kt * 32, lane), b_frag_sw(Bs, kt, wave, lane), c);
  const int n = lane & 15, mb = (lane < 16) ? 0 : 8;
  const float bv = b1[col0 + n];
#pragma unroll
  for (int r = 0; r < 8; ++r) hs[mb + r][col0 + n] = c[r] + bv;
  __syncthreads();
  if (threadIdx.x < 16) {
    float s = 0.f;
    for (int j = 0; j < 128; ++j) s += hs[threadIdx.x][j];
    const float mu = s * (1.0f / 128.0f);
    float vv = 0.f;
    for (int j = 0; j < 128; ++j) { const float t = hs[threadIdx.x][j] - mu; vv += t * t; }
    smu[threadIdx.x] = mu;
    srstd[threadIdx.x] = rsqrtf(vv * (1.0f / 128.0f) + 1e-5f);
  }
  __syncthreads();
  for (int t = threadIdx.x; t < 16 * 128; t += 256) {
    const int rr = t >> 7, cc = t & 127;
    const float val = (hs[rr][cc] - smu[rr]) * srstd[rr] * ln_g[cc] + ln_b[cc];
    hbuf[(size_t)(row0 + rr) * 128 + cc] = fmaxf(val, 0.0f);
  }
}

extern "C" void kernel_launch(void* const* d_in, const int* in_sizes, int n_in,
                              void* d_out, int out_size, void* d_ws, size_t ws_size,
                              hipStream_t stream) {
  const float* x   = (const float*)d_in[0];
  const float* ef  = (const float*)d_in[1];
  const int* eidx  = (const int*)d_in[2];
  const float* Wq  = (const float*)d_in[3];
  const float* Wk  = (const float*)d_in[4];
  const float* Wv  = (const float*)d_in[5];
  const float* We0 = (const float*)d_in[6];
  const float* We1 = (const float*)d_in[7];
  const float* W1  = (const float*)d_in[8];
  const float* b1  = (const float*)d_in[9];
  const float* lng = (const float*)d_in[10];
  const float* lnb = (const float*)d_in[11];
  const float* W2  = (const float*)d_in[12];
  const float* b2  = (const float*)d_in[13];
  float* out = (float*)d_out;

  // workspace layout (fp32 words): q,k,v [N,128] | alpha/expa [E,16] |
  // amaxkey [N,16] u32 | denom [N,16] | attn-out [N,128] | hbuf [N,128]
  float* q        = (float*)d_ws;
  float* kk       = q + (size_t)NN * 128;
  float* v        = kk + (size_t)NN * 128;
  float* alpha    = v + (size_t)NN * 128;
  unsigned* amaxk = (unsigned*)(alpha + (size_t)EE * 16);
  float* denom    = (float*)(amaxk + (size_t)NN * 16);
  float* outb     = denom + (size_t)NN * 16;
  float* hbuf     = outb + (size_t)NN * 128;

  // zero the atomic accumulators (amaxkey key(−inf)≈0 is below all real keys)
  hipMemsetAsync(amaxk, 0, (size_t)(NN * 16 * 2 + NN * 128) * sizeof(float), stream);

  gemm_k128<false><<<NN / 16, 256, 0, stream>>>(x, Wq, nullptr, q);
  gemm_k128<false><<<NN / 16, 256, 0, stream>>>(x, Wk, nullptr, kk);
  gemm_k128<false><<<NN / 16, 256, 0, stream>>>(x, Wv, nullptr, v);

  edge_alpha_kernel<<<EE / 16, 256, 0, stream>>>(ef, eidx, We0, q, kk, alpha, amaxk);
  softmax_denom_kernel<<<(EE * 16) / 256, 256, 0, stream>>>(eidx, amaxk, alpha, denom);
  edge_msg_kernel<<<EE / 16, 256, 0, stream>>>(ef, eidx, We1, v, alpha, denom, outb);

  mlp1_kernel<<<NN / 16, 256, 0, stream>>>(outb, x, W1, b1, lng, lnb, hbuf);
  gemm_k128<true><<<NN / 16, 256, 0, stream>>>(hbuf, W2, b2, out);
}
